// TimeSeriesAttention_30975304139007
// MI455X (gfx1250) — matrix operand order repaired and checked by reading, not compiled, not run on hardware
//
#include <hip/hip_runtime.h>

#define DEVINL __device__ __forceinline__

typedef __attribute__((ext_vector_type(16))) _Float16 v16h;
typedef __attribute__((ext_vector_type(8)))  _Float16 v8h;
typedef __attribute__((ext_vector_type(8)))  float    v8f;
typedef __attribute__((ext_vector_type(4)))  unsigned v4u;
typedef __attribute__((ext_vector_type(8)))  int      v8i;
typedef __attribute__((ext_vector_type(4)))  int      v4i;

constexpr int B_ = 4, T_ = 2048, C_ = 1024, H_ = 16, D_ = 64;
constexpr int BT = B_ * T_;   // 8192
constexpr int HD = H_ * D_;   // 1024

// ---------------------------------------------------------------------------
// WMMA helpers (CDNA5: D = A(16x32 f16) x B(32x16 f16) + C(16x16 f32))
// ---------------------------------------------------------------------------
DEVINL v8f wmma32f16(v16h a, v16h b, v8f c) {
  return __builtin_amdgcn_wmma_f32_16x16x32_f16(
      /*neg_a=*/false, a, /*neg_b=*/false, b,
      /*c_mod=*/(short)0, c, /*reuse_a=*/false, /*reuse_b=*/false);
}

DEVINL v8f zero8() { v8f z = {0.f,0.f,0.f,0.f,0.f,0.f,0.f,0.f}; return z; }

// A-matrix fragment (16-bit, 16x32): lane<16 holds K=0..15, lane>=16
// holds K=16..31 (contiguous).  Caller passes rowptr + (lane>>4)*16.
DEVINL v16h load_a_frag(const _Float16* p) {
  v16h r;
  ((float4*)&r)[0] = *(const float4*)(p);
  ((float4*)&r)[1] = *(const float4*)(p + 8);
  return r;
}
// B-matrix fragment (16-bit, 32x16): lanes 0-15 hold K=0..15, lanes 16-31 hold
// K=16..31 (contiguous).  Caller passes colptr + (lane>>4)*16.
DEVINL v16h load_b_frag(const _Float16* p) {
  v16h r;
  ((float4*)&r)[0] = *(const float4*)(p);
  ((float4*)&r)[1] = *(const float4*)(p + 8);
  return r;
}
DEVINL void store8h(_Float16* dst, v8f a) {
  v8h h;
#pragma unroll
  for (int i = 0; i < 8; ++i) h[i] = (_Float16)a[i];
  *(float4*)dst = *(float4*)&h;
}
DEVINL unsigned pkh(float a, float b) {
  unsigned short ua = __builtin_bit_cast(unsigned short, (_Float16)a);
  unsigned short ub = __builtin_bit_cast(unsigned short, (_Float16)b);
  return (unsigned)ua | ((unsigned)ub << 16);
}

// ---------------------------------------------------------------------------
// Tensor Data Mover: DMA a tile of 2-byte elements into LDS.
// D# layout per CDNA5 ISA.
//   address = global_addr + 2*(x + y*stride0 + z*stride1)
//   LDS fill order: x, then y, then z (contiguous).
// ---------------------------------------------------------------------------
DEVINL void tdm_load_2b(unsigned lds_off, const void* gptr,
                        unsigned tile_d0, unsigned tile_d1, unsigned tile_d2,
                        unsigned long long stride0, unsigned long long stride1,
                        unsigned tensor_d2) {
  unsigned long long ga = (unsigned long long)gptr;
  v4u g0;
  g0[0] = 1u;                                   // count=1, user mode, no gather
  g0[1] = lds_off;                              // lds_addr (bytes)
  g0[2] = (unsigned)ga;                         // global_addr[31:0]
  g0[3] = (unsigned)((ga >> 32) & 0x01FFFFFFu)  // global_addr[56:32]
          | (2u << 30);                         // type=2 ("image")
  v8i g1;
  g1[0] = (int)(1u << 16);                      // data_size=1 -> 2 bytes
  g1[1] = (int)((tile_d0 & 0xFFFFu) << 16);     // tensor_dim0 = tile_d0
  g1[2] = (int)(((tile_d0 >> 16) & 0xFFFFu) |
                ((tile_d1 & 0xFFFFu) << 16));   // tensor_dim1 = tile_d1
  g1[3] = (int)(((tile_d1 >> 16) & 0xFFFFu) |
                ((tile_d0 & 0xFFFFu) << 16));   // tile_dim0
  g1[4] = (int)((tile_d1 & 0xFFFFu) |
                ((tile_d2 & 0xFFFFu) << 16));   // tile_dim1 | tile_dim2
  g1[5] = (int)(unsigned)(stride0 & 0xFFFFFFFFull);
  g1[6] = (int)(unsigned)(((stride0 >> 32) & 0xFFFFull) |
                          ((stride1 & 0xFFFFull) << 16));
  g1[7] = (int)(unsigned)((stride1 >> 16) & 0xFFFFFFFFull);
  v4i g2;
  g2[0] = (int)tensor_d2;                       // tensor_dim2 (3D); 0 if 2D
  g2[1] = 0; g2[2] = 0; g2[3] = 0;
  v4i g3; g3[0] = 0; g3[1] = 0; g3[2] = 0; g3[3] = 0;
#if defined(__clang_major__) && __clang_major__ >= 23
  v8i g4; g4[0]=0; g4[1]=0; g4[2]=0; g4[3]=0; g4[4]=0; g4[5]=0; g4[6]=0; g4[7]=0;
  __builtin_amdgcn_tensor_load_to_lds(g0, g1, g2, g3, g4, 0);
#else
  __builtin_amdgcn_tensor_load_to_lds(g0, g1, g2, g3, 0);
#endif
}

// ---------------------------------------------------------------------------
// Precision conversion / layout kernels
// ---------------------------------------------------------------------------
__global__ void cvt_f16_kernel(const float* __restrict__ in,
                               _Float16* __restrict__ out, int n) {
  int i = blockIdx.x * 256 + threadIdx.x;
  if (i < n) out[i] = (_Float16)in[i];
}

// wq/wk/wv [H,C,D] f32  ->  wT [3][H][D][C] f16 (K-contiguous for A frags)
__global__ void cvt_wT_kernel(const float* __restrict__ wq,
                              const float* __restrict__ wk,
                              const float* __restrict__ wv,
                              _Float16* __restrict__ outT) {
  int i = blockIdx.x * 256 + threadIdx.x;
  const int hdc = H_ * D_ * C_;
  if (i >= 3 * hdc) return;
  int m = i / hdc, r = i % hdc;
  int c = r & (C_ - 1);
  int hd = r >> 10;             // r / C_
  int h = hd >> 6, d = hd & 63;
  const float* src = (m == 0) ? wq : (m == 1) ? wk : wv;
  outT[i] = (_Float16)src[((size_t)h * C_ + c) * D_ + d];
}

// w_proj [HD,C] f32 -> wpT [C][HD] f16
__global__ void cvt_wpT_kernel(const float* __restrict__ wp,
                               _Float16* __restrict__ outT) {
  int i = blockIdx.x * 256 + threadIdx.x;
  if (i >= C_ * HD) return;
  int k = i & (HD - 1), c = i >> 10;
  outT[i] = (_Float16)wp[(size_t)k * C_ + c];
}

// ---------------------------------------------------------------------------
// QKV projection:  q/k [B,H,T,D] f16,  v transposed [B,H,D,T] f16
// Block: 128 threads (4 waves), 64 rows of flattened [B*T], one head.
// Weights staged via TDM (3-D tile: 32c x 64d x 3 matrices), double-buffered.
// ---------------------------------------------------------------------------
__global__ __launch_bounds__(128) void qkv_kernel(
    const _Float16* __restrict__ xh, const _Float16* __restrict__ wT,
    _Float16* __restrict__ qh, _Float16* __restrict__ kh,
    _Float16* __restrict__ vt) {
  constexpr int WBUF = 3 * 64 * 32;             // elements per buffer
  __shared__ _Float16 wlds[2][WBUF];            // 24 KB double buffer
  const int tid = threadIdx.x;
  const int wwv = tid >> 5;
  const int lane = tid & 31, ln = lane & 15, hi = lane >> 4;
  const int h = blockIdx.y;
  const int row0 = blockIdx.x * 64;
  const int tg = row0 + wwv * 16 + ln;          // flattened B*T row (N col)
  const _Float16* xrow = xh + (size_t)tg * C_;
  const _Float16* wbase = wT + (size_t)h * D_ * C_;   // + m*H*D*C + d*C + c
  const unsigned lds0 = (unsigned)(unsigned long long)(void*)&wlds[0][0];

  v8f acc[3][4];
#pragma unroll
  for (int m = 0; m < 3; ++m)
#pragma unroll
    for (int dt = 0; dt < 4; ++dt) acc[m][dt] = zero8();

  // Prologue: DMA first weight block into buffer 0.
  if (wwv == 0)
    tdm_load_2b(lds0, wbase, /*x*/32, /*y*/64, /*z*/3,
                /*stride0=*/C_, /*stride1=*/(unsigned long long)H_ * D_ * C_,
                /*tensor_d2=*/3);

  int it = 0;
  for (int c0 = 0; c0 < C_; c0 += 32, ++it) {
    __builtin_amdgcn_s_wait_tensorcnt((short)0);  // wave0: block ready
    __syncthreads();                              // publish + retire old reads
    if (c0 + 32 < C_ && wwv == 0)                 // DMA next block, other buf
      tdm_load_2b(lds0 + (unsigned)(((it + 1) & 1) * WBUF * 2),
                  wbase + c0 + 32, 32, 64, 3,
                  C_, (unsigned long long)H_ * D_ * C_, 3);
    const _Float16* wb = &wlds[it & 1][0];
    v16h bx = load_b_frag(xrow + c0 + hi * 16);
#pragma unroll
    for (int m = 0; m < 3; ++m) {
#pragma unroll
      for (int dt = 0; dt < 4; ++dt) {
        v16h aw = load_a_frag(&wb[(m * 64 + dt * 16 + ln) * 32 + hi * 16]);
        acc[m][dt] = wmma32f16(aw, bx, acc[m][dt]);
      }
    }
  }

  const int b = tg >> 11, t = tg & (T_ - 1);
  const size_t bh = (size_t)b * H_ + h;
#pragma unroll
  for (int dt = 0; dt < 4; ++dt) {
    store8h(qh + (bh * T_ + t) * D_ + dt * 16 + hi * 8, acc[0][dt]);
    store8h(kh + (bh * T_ + t) * D_ + dt * 16 + hi * 8, acc[1][dt]);
#pragma unroll
    for (int i = 0; i < 8; ++i)   // V stored transposed [d][t]
      vt[(bh * D_ + dt * 16 + hi * 8 + i) * T_ + t] = (_Float16)acc[2][dt][i];
  }
}

// ---------------------------------------------------------------------------
// Flash attention: S^T = K x Q^T, online softmax, O^T += V^T x P^T.
// Block = 4 independent waves, each owns 16 queries; 32 keys per iteration.
// ---------------------------------------------------------------------------
__global__ __launch_bounds__(128) void attn_kernel(
    const _Float16* __restrict__ qh, const _Float16* __restrict__ kh,
    const _Float16* __restrict__ vt, _Float16* __restrict__ oh) {
  const int tid = threadIdx.x;
  const int wwv = tid >> 5;
  const int lane = tid & 31, ln = lane & 15, hi = lane >> 4;
  const int bh = blockIdx.y;
  const int b = bh >> 4, h = bh & (H_ - 1);
  const int q0 = blockIdx.x * 64 + wwv * 16;
  const int tq = q0 + ln;                        // this lane's query column
  const _Float16* qrow = qh + ((size_t)bh * T_ + tq) * D_;
  const v16h bq0 = load_b_frag(qrow + hi * 16);        // d 0..31
  const v16h bq1 = load_b_frag(qrow + 32 + hi * 16);   // d 32..63

  v8f accO[4];
#pragma unroll
  for (int dt = 0; dt < 4; ++dt) accO[dt] = zero8();
  float m_run = -1e30f, l_run = 0.f;
  const float scale = 0.03125f;                  // C^-0.5, quirk preserved

  const int s_end = q0 + 16;                     // wave-uniform key bound
  for (int s0 = 0; s0 < s_end; s0 += 32) {
    const _Float16* krow0 = kh + ((size_t)bh * T_ + s0 + ln) * D_;
    const _Float16* krow1 = krow0 + 16 * D_;
    __builtin_prefetch(krow0 + 32 * D_);         // next key tile

    v8f sa0 = wmma32f16(load_a_frag(krow0 + hi * 16), bq0, zero8());
    sa0 = wmma32f16(load_a_frag(krow0 + 32 + hi * 16), bq1, sa0);
    v8f sa1 = wmma32f16(load_a_frag(krow1 + hi * 16), bq0, zero8());
    sa1 = wmma32f16(load_a_frag(krow1 + 32 + hi * 16), bq1, sa1);

    // lane holds S^T[s, t=ln] with s = s0 + (tile? 16:0) + hi*8 + i
    float p0[8], p1[8];
    float mt = -1e30f;
#pragma unroll
    for (int i = 0; i < 8; ++i) {
      int sg = s0 + hi * 8 + i;
      p0[i] = (sg <= tq) ? sa0[i] * scale : -1e30f;
      p1[i] = (sg + 16 <= tq) ? sa1[i] * scale : -1e30f;
      mt = fmaxf(mt, fmaxf(p0[i], p1[i]));
    }
    mt = fmaxf(mt, __shfl_xor(mt, 16, 32));      // other 8 s-values per t
    const float m_new = fmaxf(m_run, mt);
    const float alpha = __expf(m_run - m_new);
    float rs = 0.f;
    union { v16h hv; unsigned u[8]; } bp;
#pragma unroll
    for (int p = 0; p < 4; ++p) {
      float e00 = __expf(p0[2*p]     - m_new);
      float e01 = __expf(p0[2*p + 1] - m_new);
      float e10 = __expf(p1[2*p]     - m_new);
      float e11 = __expf(p1[2*p + 1] - m_new);
      rs += e00 + e01 + e10 + e11;
      unsigned own0 = pkh(e00, e01);             // this lane's 2 probs, tile0
      unsigned own1 = pkh(e10, e11);             // tile1
      unsigned oth0 = (unsigned)__shfl_xor((int)own0, 16, 32);
      unsigned oth1 = (unsigned)__shfl_xor((int)own1, 16, 32);
      // B-frag K layout: lanes<16 need s 0..15 (tile0), lanes>=16 s 16..31
      bp.u[p]     = hi ? oth1 : own0;
      bp.u[4 + p] = hi ? own1 : oth0;
    }
    rs += __shfl_xor(rs, 16, 32);
    l_run = l_run * alpha + rs;
    m_run = m_new;

#pragma unroll
    for (int dt = 0; dt < 4; ++dt) {
#pragma unroll
      for (int i = 0; i < 8; ++i) accO[dt][i] *= alpha;
      const _Float16* vrow = vt + ((size_t)bh * D_ + dt * 16 + ln) * T_ + s0;
      accO[dt] = wmma32f16(load_a_frag(vrow + hi * 16), bp.hv, accO[dt]);
    }
  }

  const float inv = 1.f / l_run;
  _Float16* orow = oh + (((size_t)b * T_ + tq) * H_ + h) * D_;  // [B,T,H,D]
#pragma unroll
  for (int dt = 0; dt < 4; ++dt) {
    v8f r;
#pragma unroll
    for (int i = 0; i < 8; ++i) r[i] = accO[dt][i] * inv;
    store8h(orow + dt * 16 + hi * 8, r);
  }
}

// ---------------------------------------------------------------------------
// Output projection: out[t,c] = O[t,:] . w_proj[:,c] + b[c]  (A = wpT tiles
// via TDM double-buffered LDS, B = O^T columns), fp32 output.
// ---------------------------------------------------------------------------
__global__ __launch_bounds__(128) void proj_kernel(
    const _Float16* __restrict__ oh, const _Float16* __restrict__ wpT,
    const float* __restrict__ bias, float* __restrict__ out) {
  constexpr int PBUF = 64 * 32;                 // elements per buffer
  __shared__ _Float16 wlds[2][PBUF];            // 8 KB double buffer
  const int tid = threadIdx.x;
  const int wwv = tid >> 5;
  const int lane = tid & 31, ln = lane & 15, hi = lane >> 4;
  const int c0 = blockIdx.y * 64;
  const int tg = blockIdx.x * 64 + wwv * 16 + ln;
  const _Float16* orow = oh + (size_t)tg * HD;
  const _Float16* wbase = wpT + (size_t)c0 * HD;   // + c*HD + k
  const unsigned lds0 = (unsigned)(unsigned long long)(void*)&wlds[0][0];

  v8f acc[4];
#pragma unroll
  for (int ct = 0; ct < 4; ++ct) acc[ct] = zero8();

  if (wwv == 0)   // 2-D tile: 32 hd x 64 c rows, row stride HD
    tdm_load_2b(lds0, wbase, 32, 64, 0, HD, 0, 0);

  int it = 0;
  for (int k0 = 0; k0 < HD; k0 += 32, ++it) {
    __builtin_amdgcn_s_wait_tensorcnt((short)0);
    __syncthreads();
    if (k0 + 32 < HD && wwv == 0)
      tdm_load_2b(lds0 + (unsigned)(((it + 1) & 1) * PBUF * 2),
                  wbase + k0 + 32, 32, 64, 0, HD, 0, 0);
    const _Float16* wb = &wlds[it & 1][0];
    v16h bo = load_b_frag(orow + k0 + hi * 16);
#pragma unroll
    for (int ct = 0; ct < 4; ++ct) {
      v16h aw = load_a_frag(&wb[(ct * 16 + ln) * 32 + hi * 16]);
      acc[ct] = wmma32f16(aw, bo, acc[ct]);
    }
  }

  float* dst = out + (size_t)tg * C_;
#pragma unroll
  for (int ct = 0; ct < 4; ++ct) {
    int cb = c0 + ct * 16 + hi * 8;
    float4 b0 = *(const float4*)(bias + cb);
    float4 b1 = *(const float4*)(bias + cb + 4);
    float4 r0, r1;
    r0.x = acc[ct][0] + b0.x; r0.y = acc[ct][1] + b0.y;
    r0.z = acc[ct][2] + b0.z; r0.w = acc[ct][3] + b0.w;
    r1.x = acc[ct][4] + b1.x; r1.y = acc[ct][5] + b1.y;
    r1.z = acc[ct][6] + b1.z; r1.w = acc[ct][7] + b1.w;
    *(float4*)(dst + cb)     = r0;
    *(float4*)(dst + cb + 4) = r1;
  }
}

// ---------------------------------------------------------------------------
extern "C" void kernel_launch(void* const* d_in, const int* in_sizes, int n_in,
                              void* d_out, int out_size, void* d_ws,
                              size_t ws_size, hipStream_t stream) {
  (void)in_sizes; (void)n_in; (void)out_size; (void)ws_size;
  const float* x  = (const float*)d_in[0];
  const float* wq = (const float*)d_in[1];
  const float* wk = (const float*)d_in[2];
  const float* wv = (const float*)d_in[3];
  const float* wp = (const float*)d_in[4];
  const float* bp = (const float*)d_in[5];
  float* out = (float*)d_out;

  // f16 workspace carve-out (~88 MB total)
  _Float16* xh  = (_Float16*)d_ws;                        // BT*C
  _Float16* wT  = xh  + (size_t)BT * C_;                  // 3*H*D*C
  _Float16* wpT = wT  + (size_t)3 * H_ * D_ * C_;         // C*HD
  _Float16* qh  = wpT + (size_t)C_ * HD;                  // B*H*T*D
  _Float16* kh  = qh  + (size_t)B_ * H_ * T_ * D_;
  _Float16* vt  = kh  + (size_t)B_ * H_ * T_ * D_;        // [B,H,D,T]
  _Float16* oh  = vt  + (size_t)B_ * H_ * T_ * D_;        // [B,T,H,D]

  cvt_f16_kernel<<<(BT * C_ + 255) / 256, 256, 0, stream>>>(x, xh, BT * C_);
  cvt_wT_kernel<<<(3 * H_ * D_ * C_ + 255) / 256, 256, 0, stream>>>(wq, wk, wv, wT);
  cvt_wpT_kernel<<<(C_ * HD + 255) / 256, 256, 0, stream>>>(wp, wpT);

  qkv_kernel<<<dim3(BT / 64, H_), 128, 0, stream>>>(xh, wT, qh, kh, vt);
  attn_kernel<<<dim3(T_ / 64, B_ * H_), 128, 0, stream>>>(qh, kh, vt, oh);
  proj_kernel<<<dim3(BT / 64, C_ / 64), 128, 0, stream>>>(oh, wpT, bp, out);
}
